// GEdgeMask_19799799234723
// MI455X (gfx1250) — compile-verified, matrix operand order checked by reading, not run on hardware
//
#include <hip/hip_runtime.h>

// ---------------------------------------------------------------------------
// GEdgeMask pipeline for MI455X (gfx1250, wave32, WMMA).
//   V=50000, FN=100000, CIN=3, HID=256, E <= 3*FN.
// Layer-2 GEMM (K=256) runs on v_wmma_f32_16x16x32_f16 with f16 activations;
// layer-1 (K=3) stays VALU (WMMA padding 3->32 would waste 10x MACs).
// Segment-mean is commuted through the linear maps so atomics touch only
// 3-channel (pre-L1) and 2-channel (post-L2) data instead of 256-wide h.
// Round-3 change: pin all 8 A fragments live across an empty asm barrier so
// the backend cannot re-serialize load->wait0->wmma per k-step; we get one
// 16-load clause + one wait + 8 back-to-back WMMAs per tile.
// ---------------------------------------------------------------------------

#define V_N    50000
#define FN_N   100000
#define NFE    (3 * FN_N)          // 300000 face-edge slots
#define HID    256
#define TEMPR  0.1f
#define EPSV   1e-20f

#define RCHUNK 1024
#define RNB    ((NFE + RCHUNK - 1) / RCHUNK)   // radix blocks
#define HBINS  2048
#define MAXPE  (NFE / 2)
#define MAXNF  (4 * FN_N)

typedef __attribute__((ext_vector_type(16))) _Float16 v16h;
typedef __attribute__((ext_vector_type(8)))  _Float16 v8h;
typedef __attribute__((ext_vector_type(8)))  float    v8f;

__device__ __forceinline__ unsigned umin_(unsigned a, unsigned b) { return a < b ? a : b; }
__device__ __forceinline__ unsigned umax_(unsigned a, unsigned b) { return a > b ? a : b; }

// counter slots (u32): 0:E 1:PE 2:tbin 3:remain 4:num 5..8:class counts
// 9..12:class cursors 13:NF 14:NEr 15..18:class bases
// ---------------------------------------------------------------------------

__global__ void k_fill_u32(unsigned* p, unsigned v, unsigned n) {
  unsigned i = blockIdx.x * blockDim.x + threadIdx.x;
  if (i < n) p[i] = v;
}
__global__ void k_fill_f32(float* p, float v, unsigned n) {
  unsigned i = blockIdx.x * blockDim.x + threadIdx.x;
  if (i < n) p[i] = v;
}

// --- key building: key = min*V + max (fits u32: 50000^2 < 2^32) -------------
__global__ void k_build_keys(const int* __restrict__ face, unsigned* __restrict__ keys,
                             unsigned* __restrict__ vals) {
  unsigned i = blockIdx.x * blockDim.x + threadIdx.x;
  if (i >= NFE) return;
  unsigned j = i / FN_N, f = i % FN_N;
  int a, b;
  if (j == 0)      { a = face[f * 3 + 0]; b = face[f * 3 + 1]; }
  else if (j == 1) { a = face[f * 3 + 1]; b = face[f * 3 + 2]; }
  else             { a = face[f * 3 + 2]; b = face[f * 3 + 0]; }
  int lo = a < b ? a : b, hi = a < b ? b : a;
  keys[i] = (unsigned)lo * (unsigned)V_N + (unsigned)hi;
  vals[i] = i;
}

// --- LSD radix sort (4 x 8-bit passes; order among equal keys irrelevant) ---
__global__ void k_radix_hist(const unsigned* __restrict__ keys, unsigned* __restrict__ hist,
                             int shift, unsigned n) {
  __shared__ unsigned lh[256];
  for (int i = threadIdx.x; i < 256; i += blockDim.x) lh[i] = 0;
  __syncthreads();
  unsigned base = blockIdx.x * RCHUNK;
  unsigned end  = umin_(base + RCHUNK, n);
  for (unsigned i = base + threadIdx.x; i < end; i += blockDim.x)
    atomicAdd(&lh[(keys[i] >> shift) & 255u], 1u);
  __syncthreads();
  for (int i = threadIdx.x; i < 256; i += blockDim.x)
    hist[(unsigned)i * gridDim.x + blockIdx.x] = lh[i];
}

__global__ void k_scan_excl(const unsigned* __restrict__ in, unsigned* __restrict__ out,
                            unsigned n, unsigned* total) {
  __shared__ unsigned buf[256];
  __shared__ unsigned carry;
  if (threadIdx.x == 0) carry = 0;
  __syncthreads();
  for (unsigned base = 0; base < n; base += 256) {
    unsigned i = base + threadIdx.x;
    unsigned v = (i < n) ? in[i] : 0u;
    buf[threadIdx.x] = v;
    __syncthreads();
    for (int off = 1; off < 256; off <<= 1) {
      unsigned t = (threadIdx.x >= (unsigned)off) ? buf[threadIdx.x - off] : 0u;
      __syncthreads();
      buf[threadIdx.x] += t;
      __syncthreads();
    }
    if (i < n) out[i] = carry + buf[threadIdx.x] - v;
    __syncthreads();
    if (threadIdx.x == 0) carry += buf[255];
    __syncthreads();
  }
  if (threadIdx.x == 0 && total) *total = carry;
}

__global__ void k_radix_scatter(const unsigned* __restrict__ keys, const unsigned* __restrict__ vals,
                                unsigned* __restrict__ okeys, unsigned* __restrict__ ovals,
                                unsigned* __restrict__ offs, int shift, unsigned n) {
  unsigned base = blockIdx.x * RCHUNK;
  unsigned end  = umin_(base + RCHUNK, n);
  for (unsigned i = base + threadIdx.x; i < end; i += blockDim.x) {
    unsigned k   = keys[i];
    unsigned b   = (k >> shift) & 255u;
    unsigned pos = atomicAdd(&offs[b * gridDim.x + blockIdx.x], 1u);
    okeys[pos] = k;
    ovals[pos] = vals[i];
  }
}

// --- unique / inverse -------------------------------------------------------
__global__ void k_flags(const unsigned* __restrict__ skeys, unsigned* __restrict__ flags) {
  unsigned i = blockIdx.x * blockDim.x + threadIdx.x;
  if (i >= NFE) return;
  flags[i] = (i == 0) ? 1u : (skeys[i] != skeys[i - 1] ? 1u : 0u);
}

__global__ void k_unique_emit(const unsigned* __restrict__ skeys, const unsigned* __restrict__ svals,
                              const unsigned* __restrict__ exc, unsigned* __restrict__ ukeys,
                              unsigned* __restrict__ inv, int* __restrict__ out_edge_node) {
  unsigned i = blockIdx.x * blockDim.x + threadIdx.x;
  if (i >= NFE) return;
  unsigned flag = (i == 0) ? 1u : (skeys[i] != skeys[i - 1] ? 1u : 0u);
  unsigned uid  = exc[i] + flag - 1u;
  if (flag) {
    unsigned k = skeys[i];
    ukeys[uid] = k;
    out_edge_node[uid * 2 + 0] = (int)(k / V_N);
    out_edge_node[uid * 2 + 1] = (int)(k % V_N);
  }
  inv[svals[i]] = uid;
}

__global__ void k_num(const int* __restrict__ divp, unsigned* __restrict__ cn) {
  if (threadIdx.x || blockIdx.x) return;
  int d = divp[0]; if (d < 1) d = 1;
  cn[4] = cn[0] / (unsigned)d;
  cn[5] = cn[6] = cn[7] = cn[8] = 0u;
}

// --- line-graph connectivity ------------------------------------------------
__global__ void k_face_edges(const unsigned* __restrict__ inv, unsigned* __restrict__ esrc,
                             unsigned* __restrict__ edst) {
  unsigned f = blockIdx.x * blockDim.x + threadIdx.x;
  if (f >= FN_N) return;
  unsigned e0 = inv[f], e1 = inv[FN_N + f], e2 = inv[2 * FN_N + f];
  unsigned b = f * 6;
  esrc[b + 0] = e0; edst[b + 0] = e1;
  esrc[b + 1] = e1; edst[b + 1] = e0;
  esrc[b + 2] = e1; edst[b + 2] = e2;
  esrc[b + 3] = e2; edst[b + 3] = e1;
  esrc[b + 4] = e2; edst[b + 4] = e0;
  esrc[b + 5] = e0; edst[b + 5] = e2;
}

__global__ void k_self_loops(unsigned* __restrict__ esrc, unsigned* __restrict__ edst,
                             const unsigned* __restrict__ cn) {
  unsigned e = blockIdx.x * blockDim.x + threadIdx.x;
  if (e >= cn[0]) return;
  esrc[6 * FN_N + e] = e;
  edst[6 * FN_N + e] = e;
}

// --- edge-node features, 3-channel aggregation ------------------------------
__global__ void k_xe3(const float* __restrict__ x, const unsigned* __restrict__ ukeys,
                      float* __restrict__ xe3, const unsigned* __restrict__ cn) {
  unsigned e = blockIdx.x * blockDim.x + threadIdx.x;
  if (e >= cn[0]) return;
  unsigned k = ukeys[e], u = k / V_N, v = k % V_N;
  xe3[e * 3 + 0] = 0.5f * (x[u * 3 + 0] + x[v * 3 + 0]);
  xe3[e * 3 + 1] = 0.5f * (x[u * 3 + 1] + x[v * 3 + 1]);
  xe3[e * 3 + 2] = 0.5f * (x[u * 3 + 2] + x[v * 3 + 2]);
}

__global__ void k_agg3(const float* __restrict__ xe3, const unsigned* __restrict__ esrc,
                       const unsigned* __restrict__ edst, float* __restrict__ agg3,
                       unsigned* __restrict__ deg, const unsigned* __restrict__ cn) {
  unsigned n = 6u * FN_N + cn[0];
  unsigned i = blockIdx.x * blockDim.x + threadIdx.x;
  if (i >= n) return;
  unsigned s = esrc[i], d = edst[i];
  atomicAdd(&agg3[d * 3 + 0], xe3[s * 3 + 0]);
  atomicAdd(&agg3[d * 3 + 1], xe3[s * 3 + 1]);
  atomicAdd(&agg3[d * 3 + 2], xe3[s * 3 + 2]);
  atomicAdd(&deg[d], 1u);
}

__global__ void k_agg3_div(float* __restrict__ agg3, const unsigned* __restrict__ deg,
                           const unsigned* __restrict__ cn) {
  unsigned e = blockIdx.x * blockDim.x + threadIdx.x;
  if (e >= cn[0]) return;
  float dg = (float)umax_(deg[e], 1u);
  agg3[e * 3 + 0] /= dg;
  agg3[e * 3 + 1] /= dg;
  agg3[e * 3 + 2] /= dg;
}

// --- layer 1: K=3, VALU FMA, h stored f16 (halves HBM traffic) --------------
__global__ void k_layer1(const float* __restrict__ xe3, const float* __restrict__ agg3,
                         const float* __restrict__ W1s, const float* __restrict__ W1n,
                         const float* __restrict__ b1, _Float16* __restrict__ hbuf,
                         const unsigned* __restrict__ cn) {
  __shared__ float sWs[3 * HID], sWn[3 * HID], sb[HID];
  for (int i = threadIdx.x; i < 3 * HID; i += blockDim.x) { sWs[i] = W1s[i]; sWn[i] = W1n[i]; }
  for (int i = threadIdx.x; i < HID; i += blockDim.x) sb[i] = b1[i];
  __syncthreads();
  unsigned total = cn[0] * 32u;   // 32 threads/edge, 8 channels each
  for (unsigned w = blockIdx.x * blockDim.x + threadIdx.x; w < total;
       w += gridDim.x * blockDim.x) {
    unsigned e = w >> 5;
    int c0 = (int)(w & 31u) * 8;
    float x0 = xe3[e * 3 + 0], x1 = xe3[e * 3 + 1], x2 = xe3[e * 3 + 2];
    float a0 = agg3[e * 3 + 0], a1 = agg3[e * 3 + 1], a2 = agg3[e * 3 + 2];
    _Float16 o[8] __attribute__((aligned(16)));
#pragma unroll
    for (int i = 0; i < 8; ++i) {
      int ch = c0 + i;
      float v = sb[ch] + x0 * sWs[ch] + x1 * sWs[HID + ch] + x2 * sWs[2 * HID + ch]
                       + a0 * sWn[ch] + a1 * sWn[HID + ch] + a2 * sWn[2 * HID + ch];
      o[i] = (_Float16)fmaxf(v, 0.0f);
    }
    *(v8h*)(hbuf + (size_t)e * HID + c0) = *(const v8h*)o;
  }
}

// --- layer 2: [E,256] @ [256,4pad16] on v_wmma_f32_16x16x32_f16 -------------
// One wave per 16-edge tile; B fragments (8 k-steps) live in registers for
// the whole grid-stride loop. All 8 A fragments are pinned live across an
// asm barrier so the scheduler emits one 16-load clause + one wait + 8
// back-to-back WMMAs instead of 8 serialized load->wait0->wmma steps.
// The raw B panel is staged global->LDS with CDNA5 async copies (ASYNCcnt).
__global__ void k_layer2_wmma(const _Float16* __restrict__ hbuf,
                              const float* __restrict__ W2s, const float* __restrict__ W2n,
                              float* __restrict__ pg, const unsigned* __restrict__ cn) {
  __shared__ float Wraw[1024] __attribute__((aligned(16)));  // 512 f32 W2s | 512 f32 W2n
  __shared__ _Float16 Bs[HID * 16];  // padded B: [K=256][N=16], cols 0,1=W2s 2,3=W2n
  int tid = threadIdx.x;

  // Stage W2s|W2n (2 x 2KB) into LDS via async global->LDS 16B copies (wave 0).
  if (tid < 32) {
    unsigned lane = (unsigned)tid;
#pragma unroll
    for (int it = 0; it < 4; ++it) {
      unsigned idx = (unsigned)it * 32u + lane;  // 16B chunk id (0..127 per tensor)
      unsigned lds0 = (unsigned)(uintptr_t)(&Wraw[0]) + idx * 16u;
      unsigned long long gs =
          (unsigned long long)(uintptr_t)W2s + (unsigned long long)idx * 16u;
      asm volatile("global_load_async_to_lds_b128 %0, %1, off"
                   :: "v"(lds0), "v"(gs) : "memory");
      unsigned lds1 = (unsigned)(uintptr_t)(&Wraw[512]) + idx * 16u;
      unsigned long long gn =
          (unsigned long long)(uintptr_t)W2n + (unsigned long long)idx * 16u;
      asm volatile("global_load_async_to_lds_b128 %0, %1, off"
                   :: "v"(lds1), "v"(gn) : "memory");
    }
    asm volatile("s_wait_asynccnt 0x0" ::: "memory");
  }
  __syncthreads();

  // Expand raw f32 panel to padded f16 fragment panel in LDS.
  for (int i = tid; i < HID * 16; i += blockDim.x) {
    int k = i >> 4, n = i & 15;
    float v = 0.0f;
    if (n < 2) v = Wraw[k * 2 + n];
    else if (n < 4) v = Wraw[512 + k * 2 + (n - 2)];
    Bs[i] = (_Float16)v;
  }
  __syncthreads();

  unsigned E = cn[0];
  if (E == 0) return;
  unsigned ntiles = (E + 15u) >> 4;
  int lane = tid & 31, hf = lane >> 4, l15 = lane & 15;

  // B fragment layout (16-bit, 32x16): lane = N; per-lane halves j<4 -> K=2j(+1),
  // j>=4 -> K=16+2(j-4)(+1); lanes 16-31 shifted by +8 in K.
  v16h bf[8];
#pragma unroll
  for (int ks = 0; ks < 8; ++ks) {
#pragma unroll
    for (int m = 0; m < 16; ++m) {
      int j = m >> 1, lo = m & 1;
      int kb = (j < 4) ? (2 * j) : (16 + 2 * (j - 4));
      int K = ks * 32 + kb + lo + (hf ? 8 : 0);
      bf[ks][m] = Bs[K * 16 + l15];
    }
  }

  unsigned wid  = (blockIdx.x * blockDim.x + (unsigned)tid) >> 5;
  unsigned wtot = (gridDim.x * blockDim.x) >> 5;
  for (unsigned t = wid; t < ntiles; t += wtot) {
    unsigned row = t * 16u + (unsigned)l15;          // A row (M = l15 for both halves)
    const _Float16* hp = hbuf + (size_t)umin_(row, E - 1u) * HID + hf * 8;
    __builtin_prefetch(hbuf + (size_t)umin_(row + 16u * wtot, E - 1u) * HID, 0, 1);

    // Preload ALL 8 A fragments (one clause of 16 b128 loads).
    v16h a0, a1, a2, a3, a4, a5, a6, a7;
#define LOADFRAG(i, dst)                                                      \
    {                                                                         \
      v8h lo8 = *(const v8h*)(hp + (i) * 32);                                 \
      v8h hi8 = *(const v8h*)(hp + (i) * 32 + 16);                            \
      dst = __builtin_shufflevector(lo8, hi8, 0, 1, 2, 3, 4, 5, 6, 7,         \
                                    8, 9, 10, 11, 12, 13, 14, 15);            \
    }
    LOADFRAG(0, a0) LOADFRAG(1, a1) LOADFRAG(2, a2) LOADFRAG(3, a3)
    LOADFRAG(4, a4) LOADFRAG(5, a5) LOADFRAG(6, a6) LOADFRAG(7, a7)
#undef LOADFRAG
    // Pin all fragments live here: stops the scheduler from re-serializing
    // into per-k-step load->wait0->wmma.
    asm volatile("" : "+v"(a0), "+v"(a1), "+v"(a2), "+v"(a3),
                      "+v"(a4), "+v"(a5), "+v"(a6), "+v"(a7));

    v8f acc = {};
    acc = __builtin_amdgcn_wmma_f32_16x16x32_f16(false, a0, false, bf[0], (short)0, acc, false, false);
    acc = __builtin_amdgcn_wmma_f32_16x16x32_f16(false, a1, false, bf[1], (short)0, acc, false, false);
    acc = __builtin_amdgcn_wmma_f32_16x16x32_f16(false, a2, false, bf[2], (short)0, acc, false, false);
    acc = __builtin_amdgcn_wmma_f32_16x16x32_f16(false, a3, false, bf[3], (short)0, acc, false, false);
    acc = __builtin_amdgcn_wmma_f32_16x16x32_f16(false, a4, false, bf[4], (short)0, acc, false, false);
    acc = __builtin_amdgcn_wmma_f32_16x16x32_f16(false, a5, false, bf[5], (short)0, acc, false, false);
    acc = __builtin_amdgcn_wmma_f32_16x16x32_f16(false, a6, false, bf[6], (short)0, acc, false, false);
    acc = __builtin_amdgcn_wmma_f32_16x16x32_f16(false, a7, false, bf[7], (short)0, acc, false, false);

    // D layout: VGPR r -> (M=r, N=lane) lanes 0-15 ; (M=8+r, N=lane-16) lanes 16-31
    if (l15 < 4) {
      unsigned mbase = t * 16u + (unsigned)(hf * 8);
      float* outp = pg + (size_t)mbase * 4 + l15;
      if (mbase + 8u <= E) {                          // fast path: full tile
#pragma unroll
        for (int r = 0; r < 8; ++r) outp[(size_t)r * 4] = acc[r];
      } else {
#pragma unroll
        for (int r = 0; r < 8; ++r)
          if (mbase + (unsigned)r < E) outp[(size_t)r * 4] = acc[r];
      }
    }
  }
}

// --- 2-channel aggregation of g = h@W2n, logits, gumbel y -------------------
__global__ void k_aggg(const float* __restrict__ pg, const unsigned* __restrict__ esrc,
                       const unsigned* __restrict__ edst, float* __restrict__ aggg,
                       const unsigned* __restrict__ cn) {
  unsigned n = 6u * FN_N + cn[0];
  unsigned i = blockIdx.x * blockDim.x + threadIdx.x;
  if (i >= n) return;
  unsigned s = esrc[i], d = edst[i];
  atomicAdd(&aggg[d * 2 + 0], pg[s * 4 + 2]);
  atomicAdd(&aggg[d * 2 + 1], pg[s * 4 + 3]);
}

__device__ __forceinline__ float hash_unif(unsigned a, unsigned b) {
  unsigned x = a * 0x9E3779B9u ^ (b + 0x85EBCA6Bu);
  x ^= x >> 16; x *= 0x7FEB352Du; x ^= x >> 15; x *= 0x846CA68Bu; x ^= x >> 16;
  return ((float)(x >> 8) + 0.5f) * (1.0f / 16777216.0f);
}

__global__ void k_logits_y(const float* __restrict__ pg, const float* __restrict__ aggg,
                           const unsigned* __restrict__ deg, const float* __restrict__ b2,
                           float* __restrict__ y, const unsigned* __restrict__ cn) {
  unsigned e = blockIdx.x * blockDim.x + threadIdx.x;
  if (e >= cn[0]) return;
  float dg = (float)umax_(deg[e], 1u);
  float l0 = pg[e * 4 + 0] + aggg[e * 2 + 0] / dg + b2[0];
  float l1 = pg[e * 4 + 1] + aggg[e * 2 + 1] / dg + b2[1];
  float o0 = 1.0f / (1.0f + expf(-l0));
  float o1 = 1.0f / (1.0f + expf(-l1));
  float g0 = -logf(-logf(hash_unif(e, 0u) + EPSV) + EPSV);
  float g1 = -logf(-logf(hash_unif(e, 1u) + EPSV) + EPSV);
  float d  = ((o0 + g0) - (o1 + g1)) / TEMPR;   // 2-way softmax col 0 == sigmoid(diff)
  y[e] = 1.0f / (1.0f + expf(-d));
}

// --- deterministic top-k via histogram threshold + rank scan ----------------
__device__ __forceinline__ int ybin(float yv) {
  int b = (int)(yv * (float)HBINS);
  if (b < 0) b = 0;
  if (b > HBINS - 1) b = HBINS - 1;
  return b;
}

__global__ void k_yhist(const float* __restrict__ y, unsigned* __restrict__ hist,
                        const unsigned* __restrict__ cn) {
  unsigned e = blockIdx.x * blockDim.x + threadIdx.x;
  if (e >= cn[0]) return;
  atomicAdd(&hist[ybin(y[e])], 1u);
}

__global__ void k_thresh(const unsigned* __restrict__ hist, unsigned* __restrict__ cn) {
  if (threadIdx.x || blockIdx.x) return;
  unsigned num = cn[4], acc = 0;
  int b;
  for (b = HBINS - 1; b >= 0; --b) { acc += hist[b]; if (acc >= num) break; }
  if (b < 0) b = 0;
  cn[2] = (unsigned)b;
  cn[3] = num - (acc - hist[b]);
}

__global__ void k_bflag(const float* __restrict__ y, unsigned* __restrict__ bflag,
                        const unsigned* __restrict__ cn) {
  unsigned e = blockIdx.x * blockDim.x + threadIdx.x;
  if (e >= NFE) return;
  unsigned fl = 0;
  if (e < cn[0]) fl = (ybin(y[e]) == (int)cn[2]) ? 1u : 0u;
  bflag[e] = fl;
}

__global__ void k_select(const float* __restrict__ y, const unsigned* __restrict__ brank,
                         unsigned* __restrict__ hard, const unsigned* __restrict__ cn) {
  unsigned e = blockIdx.x * blockDim.x + threadIdx.x;
  if (e >= NFE) return;
  unsigned hv = 0;
  if (e < cn[0]) {
    int b = ybin(y[e]);
    if (b > (int)cn[2]) hv = 1u;
    else if (b == (int)cn[2] && brank[e] < cn[3]) hv = 1u;  // deterministic ties
  }
  hard[e] = hv;
}

__global__ void k_pks(const unsigned* __restrict__ ukeys, const unsigned* __restrict__ hard,
                      const unsigned* __restrict__ rank, unsigned* __restrict__ pks,
                      const unsigned* __restrict__ cn) {
  unsigned e = blockIdx.x * blockDim.x + threadIdx.x;
  if (e >= cn[0]) return;
  if (hard[e]) pks[rank[e]] = ukeys[e];   // ukeys sorted -> pks sorted
}

// --- mesh collapse ----------------------------------------------------------
__device__ __forceinline__ int bsearch_key(const unsigned* pks, unsigned PE, unsigned key) {
  unsigned lo = 0, hi = PE;
  while (lo < hi) { unsigned mid = (lo + hi) >> 1; if (pks[mid] < key) lo = mid + 1; else hi = mid; }
  if (lo < PE && pks[lo] == key) return (int)lo;
  return -1;
}

__device__ __forceinline__ unsigned fekey(const int* face, unsigned f, int j) {
  int a, b;
  if (j == 0)      { a = face[f * 3 + 0]; b = face[f * 3 + 1]; }
  else if (j == 1) { a = face[f * 3 + 1]; b = face[f * 3 + 2]; }
  else             { a = face[f * 3 + 2]; b = face[f * 3 + 0]; }
  int lo = a < b ? a : b, hi = a < b ? b : a;
  return (unsigned)lo * (unsigned)V_N + (unsigned)hi;
}

__global__ void k_class_count(const int* __restrict__ face, const unsigned* __restrict__ pks,
                              unsigned* __restrict__ cn) {
  unsigned f = blockIdx.x * blockDim.x + threadIdx.x;
  if (f >= FN_N) return;
  unsigned PE = cn[1];
  int pen = 0;
  for (int j = 0; j < 3; ++j)
    if (bsearch_key(pks, PE, fekey(face, f, j)) >= 0) pen++;
  atomicAdd(&cn[5 + pen], 1u);
}

__global__ void k_offsets(unsigned* __restrict__ cn) {
  if (threadIdx.x || blockIdx.x) return;
  unsigned n0 = cn[5], n1 = cn[6], n2 = cn[7], n3 = cn[8];
  cn[15] = 0u;
  cn[16] = n0;
  cn[17] = n0 + 2u * n1;
  cn[18] = n0 + 2u * n1 + 3u * n2;
  cn[13] = n0 + 2u * n1 + 3u * n2 + 4u * n3;      // NF
  cn[14] = (unsigned)V_N + cn[1] + 3u * cn[13];   // NEr (self pairs + 3 per face)
  cn[9] = cn[10] = cn[11] = cn[12] = 0u;          // class cursors
}

__device__ __forceinline__ void roll6(const int* vv, int r, int* w) {
  // jnp.roll(vv, r): w[k] = vv[(k - r) mod 6]
  for (int k = 0; k < 6; ++k) w[k] = vv[((k - r) % 6 + 6) % 6];
}

__global__ void k_emit_faces(const int* __restrict__ face, const unsigned* __restrict__ pks,
                             unsigned* __restrict__ cn, int* __restrict__ d_out_i) {
  unsigned f = blockIdx.x * blockDim.x + threadIdx.x;
  if (f >= FN_N) return;
  unsigned PE = cn[1], E = cn[0];
  int pei[3], fm[3], pen = 0, psi = 0;
  for (int j = 0; j < 3; ++j) {
    int p = bsearch_key(pks, PE, fekey(face, f, (int)j));
    fm[j] = (p >= 0);
    pei[j] = (p >= 0) ? (int)(V_N + (unsigned)p) : -1;
    if (fm[j]) { pen++; psi += j; }
  }
  int vv[6] = { face[f * 3 + 0], pei[0], face[f * 3 + 1], pei[1], face[f * 3 + 2], pei[2] };
  unsigned slot = atomicAdd(&cn[9 + pen], 1u);
  int* nf = d_out_i + (size_t)2 * E;   // new_faces region
  int w[6];
  if (pen == 0) {
    unsigned b = cn[15] + slot;
    nf[b * 3 + 0] = vv[0]; nf[b * 3 + 1] = vv[2]; nf[b * 3 + 2] = vv[4];
  } else if (pen == 1) {
    int r = (psi == 0) ? 0 : (psi == 1 ? -2 : 2);
    roll6(vv, r, w);
    unsigned b = cn[16] + slot * 2u;
    nf[b * 3 + 0] = w[0]; nf[b * 3 + 1] = w[1]; nf[b * 3 + 2] = w[4];
    b++;
    nf[b * 3 + 0] = w[1]; nf[b * 3 + 1] = w[2]; nf[b * 3 + 2] = w[4];
  } else if (pen == 2) {
    int r = (psi == 1) ? 0 : (psi == 2 ? 2 : -2);
    roll6(vv, r, w);
    unsigned b = cn[17] + slot * 3u;
    nf[b * 3 + 0] = w[0]; nf[b * 3 + 1] = w[1]; nf[b * 3 + 2] = w[4];
    b++;
    nf[b * 3 + 0] = w[1]; nf[b * 3 + 1] = w[2]; nf[b * 3 + 2] = w[3];
    b++;
    nf[b * 3 + 0] = w[1]; nf[b * 3 + 1] = w[3]; nf[b * 3 + 2] = w[4];
  } else {
    unsigned b = cn[18] + slot * 4u;
    nf[b * 3 + 0] = vv[0]; nf[b * 3 + 1] = vv[1]; nf[b * 3 + 2] = vv[5];
    b++;
    nf[b * 3 + 0] = vv[1]; nf[b * 3 + 1] = vv[2]; nf[b * 3 + 2] = vv[3];
    b++;
    nf[b * 3 + 0] = vv[1]; nf[b * 3 + 1] = vv[3]; nf[b * 3 + 2] = vv[5];
    b++;
    nf[b * 3 + 0] = vv[3]; nf[b * 3 + 1] = vv[4]; nf[b * 3 + 2] = vv[5];
  }
}

__global__ void k_new_edges(const unsigned* __restrict__ cn, int* __restrict__ d_out_i) {
  unsigned E = cn[0], PE = cn[1], NF = cn[13], NEr = cn[14];
  const int* nf = d_out_i + (size_t)2 * E;
  int* ne = d_out_i + (size_t)2 * E + (size_t)3 * NF;   // [2][NEr] row-major
  unsigned i = blockIdx.x * blockDim.x + threadIdx.x;
  unsigned nSelf = (unsigned)V_N + PE;
  if (i < nSelf) { ne[i] = (int)i; ne[NEr + i] = (int)i; }
  if (i < NF) {
    int a = nf[i * 3 + 0], b = nf[i * 3 + 1], c = nf[i * 3 + 2];
    unsigned p0 = nSelf + i, p1 = nSelf + NF + i, p2 = nSelf + 2u * NF + i;
    ne[p0] = a; ne[NEr + p0] = b;
    ne[p1] = b; ne[NEr + p1] = c;
    ne[p2] = c; ne[NEr + p2] = a;
  }
}

__global__ void k_mask(const float* __restrict__ y, const unsigned* __restrict__ hard,
                       const unsigned* __restrict__ cn, float* __restrict__ d_out_f) {
  unsigned e = blockIdx.x * blockDim.x + threadIdx.x;
  if (e >= cn[0]) return;
  size_t off = (size_t)2 * cn[0] + (size_t)3 * cn[13] + (size_t)2 * cn[14];
  float yh = hard[e] ? 1.0f : 0.0f;
  d_out_f[off + e] = (yh - y[e]) + y[e];   // straight-through value
}

// ---------------------------------------------------------------------------
extern "C" void kernel_launch(void* const* d_in, const int* in_sizes, int n_in,
                              void* d_out, int out_size, void* d_ws, size_t ws_size,
                              hipStream_t stream) {
  (void)in_sizes; (void)n_in; (void)out_size; (void)ws_size;
  const float* x    = (const float*)d_in[0];
  const int*   face = (const int*)  d_in[1];
  const int*   divp = (const int*)  d_in[4];
  const float* W1s  = (const float*)d_in[5];
  const float* W1n  = (const float*)d_in[6];
  const float* b1   = (const float*)d_in[7];
  const float* W2s  = (const float*)d_in[8];
  const float* W2n  = (const float*)d_in[9];
  const float* b2   = (const float*)d_in[10];

  char* wp = (char*)d_ws;
  size_t off = 0;
  auto alloc = [&](size_t bytes) -> char* {
    char* p = wp + off;
    off = (off + bytes + 255) & ~(size_t)255;
    return p;
  };
  unsigned*  cnt    = (unsigned*)alloc(64 * 4);
  unsigned*  keysA  = (unsigned*)alloc((size_t)NFE * 4);
  unsigned*  idxA   = (unsigned*)alloc((size_t)NFE * 4);
  unsigned*  keysB  = (unsigned*)alloc((size_t)NFE * 4);
  unsigned*  idxB   = (unsigned*)alloc((size_t)NFE * 4);
  unsigned*  histR  = (unsigned*)alloc((size_t)256 * RNB * 4);
  unsigned*  offsR  = (unsigned*)alloc((size_t)256 * RNB * 4);
  unsigned*  flags  = (unsigned*)alloc((size_t)NFE * 4);
  unsigned*  exc    = (unsigned*)alloc((size_t)NFE * 4);
  unsigned*  ukeys  = (unsigned*)alloc((size_t)NFE * 4);
  unsigned*  inv    = (unsigned*)alloc((size_t)NFE * 4);
  unsigned*  esrc   = (unsigned*)alloc((size_t)(6 * FN_N + NFE) * 4);
  unsigned*  edst   = (unsigned*)alloc((size_t)(6 * FN_N + NFE) * 4);
  unsigned*  deg    = (unsigned*)alloc((size_t)NFE * 4);
  float*     xe3    = (float*)   alloc((size_t)NFE * 3 * 4);
  float*     agg3   = (float*)   alloc((size_t)NFE * 3 * 4);
  _Float16*  hbuf   = (_Float16*)alloc((size_t)NFE * HID * 2);
  float*     pg     = (float*)   alloc((size_t)NFE * 4 * 4);
  float*     aggg   = (float*)   alloc((size_t)NFE * 2 * 4);
  float*     y      = (float*)   alloc((size_t)NFE * 4);
  unsigned*  hist2  = (unsigned*)alloc((size_t)HBINS * 4);
  unsigned*  bflag  = (unsigned*)alloc((size_t)NFE * 4);
  unsigned*  brank  = (unsigned*)alloc((size_t)NFE * 4);
  unsigned*  hard   = (unsigned*)alloc((size_t)NFE * 4);
  unsigned*  rank   = (unsigned*)alloc((size_t)NFE * 4);
  unsigned*  pks    = (unsigned*)alloc((size_t)MAXPE * 4);

  auto g = [](unsigned n) -> unsigned { return (n + 255u) / 256u; };
  const unsigned NEG = 6u * FN_N + (unsigned)NFE;  // max line-graph edges

  // 0) zero accumulators + counters (ws is NOT re-poisoned between replays)
  k_fill_u32<<<1, 64, 0, stream>>>(cnt, 0u, 64u);
  k_fill_u32<<<g(NFE), 256, 0, stream>>>(deg, 0u, NFE);
  k_fill_f32<<<g(3 * NFE), 256, 0, stream>>>(agg3, 0.0f, 3u * NFE);
  k_fill_f32<<<g(2 * NFE), 256, 0, stream>>>(aggg, 0.0f, 2u * NFE);
  k_fill_u32<<<g(HBINS), 256, 0, stream>>>(hist2, 0u, HBINS);

  // 1) keys + radix sort (u32, 4 passes)
  k_build_keys<<<g(NFE), 256, 0, stream>>>(face, keysA, idxA);
  unsigned *ka = keysA, *va = idxA, *kb = keysB, *vb = idxB;
  for (int p = 0; p < 4; ++p) {
    int shift = p * 8;
    k_radix_hist<<<RNB, 256, 0, stream>>>(ka, histR, shift, NFE);
    k_scan_excl<<<1, 256, 0, stream>>>(histR, offsR, 256u * RNB, nullptr);
    k_radix_scatter<<<RNB, 256, 0, stream>>>(ka, va, kb, vb, offsR, shift, NFE);
    unsigned* t;
    t = ka; ka = kb; kb = t;
    t = va; va = vb; vb = t;
  }
  // sorted result back in keysA/idxA (even #swaps)

  // 2) unique + inverse mapping; edge_node straight into d_out
  k_flags<<<g(NFE), 256, 0, stream>>>(keysA, flags);
  k_scan_excl<<<1, 256, 0, stream>>>(flags, exc, NFE, &cnt[0]);   // cnt[0] = E
  k_unique_emit<<<g(NFE), 256, 0, stream>>>(keysA, idxA, exc, ukeys, inv, (int*)d_out);
  k_num<<<1, 1, 0, stream>>>(divp, cnt);

  // 3) line-graph connectivity + degree + 3-channel mean-agg
  k_face_edges<<<g(FN_N), 256, 0, stream>>>(inv, esrc, edst);
  k_self_loops<<<g(NFE), 256, 0, stream>>>(esrc, edst, cnt);
  k_xe3<<<g(NFE), 256, 0, stream>>>(x, ukeys, xe3, cnt);
  k_agg3<<<g(NEG), 256, 0, stream>>>(xe3, esrc, edst, agg3, deg, cnt);
  k_agg3_div<<<g(NFE), 256, 0, stream>>>(agg3, deg, cnt);

  // 4) layer 1 (VALU, f16 out) ; layer 2 (WMMA f16->f32)
  k_layer1<<<g(NFE * 32u), 256, 0, stream>>>(xe3, agg3, W1s, W1n, b1, hbuf, cnt);
  k_layer2_wmma<<<1024, 256, 0, stream>>>(hbuf, W2s, W2n, pg, cnt);

  // 5) 2-channel agg of g, logits, gumbel y
  k_aggg<<<g(NEG), 256, 0, stream>>>(pg, esrc, edst, aggg, cnt);
  k_logits_y<<<g(NFE), 256, 0, stream>>>(pg, aggg, deg, b2, y, cnt);

  // 6) deterministic top-(E/div)
  k_yhist<<<g(NFE), 256, 0, stream>>>(y, hist2, cnt);
  k_thresh<<<1, 1, 0, stream>>>(hist2, cnt);
  k_bflag<<<g(NFE), 256, 0, stream>>>(y, bflag, cnt);
  k_scan_excl<<<1, 256, 0, stream>>>(bflag, brank, NFE, nullptr);
  k_select<<<g(NFE), 256, 0, stream>>>(y, brank, hard, cnt);
  k_scan_excl<<<1, 256, 0, stream>>>(hard, rank, NFE, &cnt[1]);   // cnt[1] = PE
  k_pks<<<g(NFE), 256, 0, stream>>>(ukeys, hard, rank, pks, cnt);

  // 7) mesh collapse -> new_faces, new_edges, mask into d_out
  k_class_count<<<g(FN_N), 256, 0, stream>>>(face, pks, cnt);
  k_offsets<<<1, 1, 0, stream>>>(cnt);
  k_emit_faces<<<g(FN_N), 256, 0, stream>>>(face, pks, cnt, (int*)d_out);
  k_new_edges<<<g(MAXNF), 256, 0, stream>>>(cnt, (int*)d_out);
  k_mask<<<g(NFE), 256, 0, stream>>>(y, hard, cnt, (float*)d_out);
}
